// TransformerLM_21500606284073
// MI455X (gfx1250) — matrix operand order repaired and checked by reading, not compiled, not run on hardware
//
#include <hip/hip_runtime.h>
#include <hip/hip_bf16.h>
#include <math.h>

// ---------------- model constants (from reference) ----------------
#define LNUM 4
#define MSEG 3
#define XLEN 512
#define BNUM 2
#define DMOD 1024
#define HNUM 16
#define DHEAD 64
#define DFFN 4096
#define VOC 32000
#define TLEN ((MSEG + 1) * XLEN)   // 2048
#define XB   (XLEN * BNUM)         // 1024 token rows
#define TB   (TLEN * BNUM)         // 4096 rows of c / kv

typedef __bf16 bf16;
typedef __attribute__((ext_vector_type(16))) __bf16 v16bf;
typedef __attribute__((ext_vector_type(8)))  __bf16 v8bf;
typedef __attribute__((ext_vector_type(8)))  float  v8f;
typedef __attribute__((ext_vector_type(4)))  int    v4i;

// pointer-to-AS-qualified-v4i types for the async DMA builtin
typedef __attribute__((address_space(1))) v4i* gv4i_p;   // global
typedef __attribute__((address_space(3))) v4i* lv4i_p;   // LDS

#if __has_builtin(__builtin_amdgcn_global_load_async_to_lds_b128)
#define USE_ASYNC_A 1
#else
#define USE_ASYNC_A 0
#endif

__device__ __forceinline__ void wait_async0() {
#if USE_ASYNC_A
#if __has_builtin(__builtin_amdgcn_s_wait_asynccnt)
  __builtin_amdgcn_s_wait_asynccnt(0);
#else
  asm volatile("s_wait_asynccnt 0x0" ::: "memory");
#endif
#endif
}

// ---------------- WMMA GEMM: C[M,N] = A[M,K] * B[K,N] (f32 in, bf16 WMMA, f32 out)
// 64x64 block tile, BK=64 (two 16x16x32 k-substeps), 4 waves each owning a 32x32
// quadrant, double-buffered LDS, one barrier + 8 WMMAs per K-step.
// Grid: blockIdx.x = M-tile (few, co-scheduled), blockIdx.y = N-tile, so all
// blocks sharing a B-slab run concurrently -> weights stream from HBM ~once
// and repeats hit the 192MB L2.
#define GBM 64
#define GBN 64
#define GBK 64
#define LDAB 72   // bf16 tile stride: 144B row (16B aligned, 36-bank skew)
#define LDAF 68   // f32 A tile stride: 272B row (16B aligned, 4-bank skew)

__device__ __forceinline__ v16bf frag_a_bf16(const bf16* rowbase, int half) {
  // 16-bit A 16x32 layout per lane (m = lane&15):
  //   e=0..7  -> K = half*8 + e ; e=8..15 -> K = 16 + half*8 + (e-8)
  v8bf lo = *(const v8bf*)(rowbase + half * 8);
  v8bf hi = *(const v8bf*)(rowbase + 16 + half * 8);
  v16bf a;
#pragma unroll
  for (int e = 0; e < 8; ++e) { a[e] = lo[e]; a[e + 8] = hi[e]; }
  return a;
}

__device__ __forceinline__ v16bf frag_a_f32(const float* rowbase, int half) {
  // same K mapping, source is f32 in LDS; converts lower to v_cvt_pk_bf16_f32
  float4 a0 = *(const float4*)(rowbase + half * 8);
  float4 a1 = *(const float4*)(rowbase + half * 8 + 4);
  float4 a2 = *(const float4*)(rowbase + 16 + half * 8);
  float4 a3 = *(const float4*)(rowbase + 16 + half * 8 + 4);
  v16bf a;
  a[0] = (bf16)a0.x;  a[1] = (bf16)a0.y;  a[2] = (bf16)a0.z;  a[3] = (bf16)a0.w;
  a[4] = (bf16)a1.x;  a[5] = (bf16)a1.y;  a[6] = (bf16)a1.z;  a[7] = (bf16)a1.w;
  a[8] = (bf16)a2.x;  a[9] = (bf16)a2.y;  a[10] = (bf16)a2.z; a[11] = (bf16)a2.w;
  a[12] = (bf16)a3.x; a[13] = (bf16)a3.y; a[14] = (bf16)a3.z; a[15] = (bf16)a3.w;
  return a;
}

__device__ __forceinline__ v16bf frag_b(const bf16* colbase, int half) {
  // B 32x16: lane n = lane&15; e=0..7 -> K = half*8 + e ; e=8..15 -> K = 16 + half*8 + (e-8)
  v8bf lo = *(const v8bf*)(colbase + half * 8);
  v8bf hi = *(const v8bf*)(colbase + 16 + half * 8);
  v16bf b;
#pragma unroll
  for (int e = 0; e < 8; ++e) { b[e] = lo[e]; b[e + 8] = hi[e]; }
  return b;
}

__device__ __forceinline__ void store_bf16x4(bf16* dst8B, float x, float y, float z, float w) {
  union { bf16 h[4]; unsigned long long u; } p;
  p.h[0] = (bf16)x; p.h[1] = (bf16)y; p.h[2] = (bf16)z; p.h[3] = (bf16)w;
  *(unsigned long long*)dst8B = p.u;   // aligned ds_store_b64
}

// EPI: 0 = plain store, 1 = +bias then relu, 2 = +bias
template <int EPI, int N, int K>
__global__ __launch_bounds__(128) void gemm_bf16_wmma(
    const float* __restrict__ A, const float* __restrict__ Bm,
    float* __restrict__ C, const float* __restrict__ bias) {
#if USE_ASYNC_A
  __shared__ __align__(16) float Asf[2][GBM * LDAF];  // f32 A tile (DMA target)
#else
  __shared__ __align__(16) bf16 As[2][GBM * LDAB];    // bf16 A tile
#endif
  __shared__ __align__(16) bf16 Bs[2][GBN * LDAB];    // Bs[buf][n][k] (transposed)
  const int t = threadIdx.x;
  const int lane = t & 31, wave = t >> 5;
  const int half = lane >> 4, l16 = lane & 15;
  const int rowBase = blockIdx.x * GBM;   // M-tile: fast grid dim (co-scheduled)
  const int colBase = blockIdx.y * GBN;   // N-tile: slow grid dim
  const int qr = (wave >> 1) * 32;
  const int qc = (wave & 1) * 32;
  constexpr int NIT = K / GBK;

  v8f acc[2][2] = {};
#if !USE_ASYNC_A
  float4 ra[8];       // A staging (fallback only)
#endif
  float  rb[8][4];    // B staging: 8 column-chunks of 4 consecutive k

  auto prefetchA = [&](int k0, int buf) {
#if USE_ASYNC_A
#pragma unroll
    for (int j = 0; j < 8; ++j) {
      int lin = j * 128 + t, r = lin >> 4, c4 = (lin & 15) << 2;
      const float* gp = A + (size_t)(rowBase + r) * K + k0 + c4;
      __builtin_amdgcn_global_load_async_to_lds_b128(
          (gv4i_p)gp, (lv4i_p)&Asf[buf][r * LDAF + c4], 0, 0);
    }
#else
    (void)buf;
#pragma unroll
    for (int j = 0; j < 8; ++j) {
      int lin = j * 128 + t, r = lin >> 4, c4 = (lin & 15) << 2;
      ra[j] = *(const float4*)(A + (size_t)(rowBase + r) * K + k0 + c4);
    }
#endif
  };
  auto loadB = [&](int k0) {
#pragma unroll
    for (int j = 0; j < 8; ++j) {
      int lin = j * 128 + t, n = lin & 63, kq = (lin >> 6) << 2;
      const float* bp = Bm + (size_t)(k0 + kq) * N + colBase + n;  // 4 coalesced b32
      rb[j][0] = bp[0];
      rb[j][1] = bp[N];
      rb[j][2] = bp[2 * N];
      rb[j][3] = bp[3 * N];
    }
  };
  auto stage = [&](int buf) {
#if !USE_ASYNC_A
#pragma unroll
    for (int j = 0; j < 8; ++j) {
      int lin = j * 128 + t, r = lin >> 4, c4 = (lin & 15) << 2;
      store_bf16x4(&As[buf][r * LDAB + c4], ra[j].x, ra[j].y, ra[j].z, ra[j].w);
    }
#endif
#pragma unroll
    for (int j = 0; j < 8; ++j) {
      int lin = j * 128 + t, n = lin & 63, kq = (lin >> 6) << 2;
      store_bf16x4(&Bs[buf][n * LDAB + kq], rb[j][0], rb[j][1], rb[j][2], rb[j][3]);
    }
  };
  auto compute = [&](int buf) {
#pragma unroll
    for (int ks = 0; ks < 2; ++ks) {
      v16bf af[2], bf_[2];
#pragma unroll
      for (int ti = 0; ti < 2; ++ti) {
#if USE_ASYNC_A
        af[ti] = frag_a_f32(&Asf[buf][(qr + ti * 16 + l16) * LDAF + ks * 32], half);
#else
        af[ti] = frag_a_bf16(&As[buf][(qr + ti * 16 + l16) * LDAB + ks * 32], half);
#endif
      }
#pragma unroll
      for (int tj = 0; tj < 2; ++tj)
        bf_[tj] = frag_b(&Bs[buf][(qc + tj * 16 + l16) * LDAB + ks * 32], half);
#pragma unroll
      for (int ti = 0; ti < 2; ++ti)
#pragma unroll
        for (int tj = 0; tj < 2; ++tj)
          acc[ti][tj] = __builtin_amdgcn_wmma_f32_16x16x32_bf16(
              false, af[ti], false, bf_[tj], (short)0, acc[ti][tj], false, false);
    }
  };

  // prologue: fill buffer 0 (async A DMA + register-staged B)
  prefetchA(0, 0);
  loadB(0);
  stage(0);
  wait_async0();
  __syncthreads();

#pragma unroll 1
  for (int it = 0; it < NIT; ++it) {
    const int buf = it & 1;
    if (it + 1 < NIT) { prefetchA((it + 1) * GBK, buf ^ 1); loadB((it + 1) * GBK); }
    compute(buf);
    if (it + 1 < NIT) stage(buf ^ 1);
    wait_async0();       // our DMA into buf^1 done before publishing barrier
    __syncthreads();     // publish buf^1; all reads of buf retired
  }

  // epilogue: C/D layout -> VGPR v holds (m = v + half*8, n = lane&15)
#pragma unroll
  for (int ti = 0; ti < 2; ++ti)
#pragma unroll
    for (int tj = 0; tj < 2; ++tj) {
      int n = colBase + qc + tj * 16 + l16;
      float bv = (EPI != 0) ? bias[n] : 0.0f;
      float* cp = C + (size_t)(rowBase + qr + ti * 16 + half * 8) * N + n;
#pragma unroll
      for (int v = 0; v < 8; ++v) {
        float x = acc[ti][tj][v];
        if (EPI != 0) x += bv;
        if (EPI == 1) x = fmaxf(x, 0.0f);
        cp[v * N] = x;   // immediate offsets (N constexpr)
      }
    }
}

// ---------------- embedding gather ----------------
__global__ __launch_bounds__(256) void embed_kernel(
    const int* __restrict__ ids, const float* __restrict__ table, float* __restrict__ h) {
  int row = blockIdx.x;                       // row = i*B + b
  int id = ids[row];
#pragma unroll
  for (int j = 0; j < 4; ++j) {
    int d = threadIdx.x + j * 256;
    h[(size_t)row * DMOD + d] = table[(size_t)id * DMOD + d];
  }
}

// ---------------- r vectors: only 4 relative positions survive the rel_shift diag ----
__global__ __launch_bounds__(256) void rel_kernel(
    const float* __restrict__ Wrel_l, float* __restrict__ r4) {
  __shared__ float pe[DMOD];
  int m = blockIdx.x;                 // 0..M
  int e = blockIdx.y * 256 + threadIdx.x;
  float pos = (float)((MSEG - m) * XLEN);   // pos_seq[(m+1)X-1] = (M-m)X
  for (int d = threadIdx.x; d < DMOD; d += 256) {
    int j = d & (DMOD / 2 - 1);
    float invf = powf(10000.0f, -(float)(2 * j) / (float)DMOD);
    float ang = pos * invf;
    pe[d] = (d < DMOD / 2) ? sinf(ang) : cosf(ang);
  }
  __syncthreads();
  float accv = 0.0f;
  for (int d = 0; d < DMOD; ++d) accv += pe[d] * Wrel_l[(size_t)d * DMOD + e];
  r4[m * DMOD + e] = accv;
}

// ---------------- diagonal-score attention + softmax over S=2 + V gather ----------
__global__ __launch_bounds__(64) void attn_kernel(
    const float* __restrict__ q, const float* __restrict__ kv,
    const float* __restrict__ r4, const float* __restrict__ ub,
    const float* __restrict__ vb, const int* __restrict__ indices,
    float* __restrict__ av) {
  __shared__ float red[64];
  const int t = threadIdx.x;                    // DH lane
  const int i = blockIdx.x >> 5;                // / (B*H)
  const int rem = blockIdx.x & 31;
  const int b = rem >> 4, n = rem & 15;
  const float scale = 0.125f;                   // 1/sqrt(64)

  float qv = q[((size_t)(i * BNUM + b)) * DMOD + n * DHEAD + t];
  float qu = qv + ub[n * DHEAD + t];
  float qb = qv + vb[n * DHEAD + t];

  float score[MSEG + 1];
#pragma unroll
  for (int m = 0; m <= MSEG; ++m) {
    size_t krow = (size_t)((m * XLEN + i) * BNUM + b);
    float kval = kv[krow * (2 * DMOD) + n * DHEAD + t];
    float rval = r4[m * DMOD + n * DHEAD + t];
    red[t] = qu * kval + qb * rval;
    __syncthreads();
    for (int off = 32; off; off >>= 1) {
      if (t < off) red[t] += red[t + off];
      __syncthreads();
    }
    score[m] = red[0] * scale;
    __syncthreads();
  }
  int s0 = indices[0 * BNUM + b];
  int s1 = indices[1 * BNUM + b];
  float e0 = score[s0], e1 = score[s1];
  float mx = fmaxf(e0, e1);
  float p0 = expf(e0 - mx), p1 = expf(e1 - mx);
  float inv = 1.0f / (p0 + p1);
  p0 *= inv; p1 *= inv;
  float v0 = kv[(size_t)((s0 * XLEN + i) * BNUM + b) * (2 * DMOD) + DMOD + n * DHEAD + t];
  float v1 = kv[(size_t)((s1 * XLEN + i) * BNUM + b) * (2 * DMOD) + DMOD + n * DHEAD + t];
  av[((size_t)(i * BNUM + b)) * DMOD + n * DHEAD + t] = p0 * v0 + p1 * v1;
}

// ---------------- residual + LayerNorm (in-place on h) ----------------
__global__ __launch_bounds__(256) void ln_residual_kernel(
    float* __restrict__ h, const float* __restrict__ res,
    const float* __restrict__ g, const float* __restrict__ b) {
  __shared__ float s1[256], s2[256];
  const int row = blockIdx.x, t = threadIdx.x;
  float x[4], lsum = 0.0f, lsq = 0.0f;
#pragma unroll
  for (int j = 0; j < 4; ++j) {
    int d = t + j * 256;
    float v = h[(size_t)row * DMOD + d] + res[(size_t)row * DMOD + d];
    x[j] = v; lsum += v; lsq += v * v;
  }
  s1[t] = lsum; s2[t] = lsq; __syncthreads();
  for (int off = 128; off; off >>= 1) {
    if (t < off) { s1[t] += s1[t + off]; s2[t] += s2[t + off]; }
    __syncthreads();
  }
  float mu = s1[0] * (1.0f / DMOD);
  float var = s2[0] * (1.0f / DMOD) - mu * mu;
  float inv = rsqrtf(var + 1e-5f);
#pragma unroll
  for (int j = 0; j < 4; ++j) {
    int d = t + j * 256;
    h[(size_t)row * DMOD + d] = (x[j] - mu) * inv * g[d] + b[d];
  }
}

// ---------------- host orchestration ----------------
extern "C" void kernel_launch(void* const* d_in, const int* in_sizes, int n_in,
                              void* d_out, int out_size, void* d_ws, size_t ws_size,
                              hipStream_t stream) {
  const int*   input_ids = (const int*)d_in[0];
  const float* memory    = (const float*)d_in[1];
  const int*   indices   = (const int*)d_in[2];
  const float* emb       = (const float*)d_in[3];
  const float* dec_w     = (const float*)d_in[4];
  const float* Wq        = (const float*)d_in[5];
  const float* Wkv       = (const float*)d_in[6];
  const float* Wrel      = (const float*)d_in[7];
  const float* Wo        = (const float*)d_in[8];
  const float* u_bias    = (const float*)d_in[9];
  const float* v_bias    = (const float*)d_in[10];
  const float* ln1g      = (const float*)d_in[11];
  const float* ln1b      = (const float*)d_in[12];
  const float* W1        = (const float*)d_in[13];
  const float* b1f       = (const float*)d_in[14];
  const float* W2        = (const float*)d_in[15];
  const float* b2f       = (const float*)d_in[16];
  const float* ln2g      = (const float*)d_in[17];
  const float* ln2b      = (const float*)d_in[18];

  // workspace carve-up (~84 MB of f32)
  float* ws = (float*)d_ws;
  size_t o = 0;
  float* h    = ws + o; o += (size_t)XB * DMOD;
  float* c    = ws + o; o += (size_t)TB * DMOD;
  float* kvb  = ws + o; o += (size_t)TB * 2 * DMOD;
  float* qb   = ws + o; o += (size_t)XB * DMOD;
  float* r4   = ws + o; o += (size_t)(MSEG + 1) * DMOD;
  float* avb  = ws + o; o += (size_t)XB * DMOD;
  float* attn = ws + o; o += (size_t)XB * DMOD;
  float* hid  = ws + o; o += (size_t)XB * DFFN;
  float* ffb  = ws + o; o += (size_t)XB * DMOD;

  embed_kernel<<<XB, 256, 0, stream>>>(input_ids, emb, h);

  const size_t memBytes = (size_t)MSEG * XLEN * BNUM * DMOD * sizeof(float);
  const size_t hBytes   = (size_t)XB * DMOD * sizeof(float);

  for (int l = 0; l < LNUM; ++l) {
    // c = concat(memory[l].reshape(M*X,B,D), h)  -- both contiguous
    (void)hipMemcpyAsync(c, memory + (size_t)l * MSEG * XLEN * BNUM * DMOD, memBytes,
                         hipMemcpyDeviceToDevice, stream);
    (void)hipMemcpyAsync(c + (size_t)MSEG * XLEN * BNUM * DMOD, h, hBytes,
                         hipMemcpyDeviceToDevice, stream);

    // kv = c @ Wkv[l]           (4096 x 2048 x 1024)   grid: (Mtiles, Ntiles)
    gemm_bf16_wmma<0, 2 * DMOD, DMOD><<<dim3(TB / GBM, (2 * DMOD) / GBN), 128, 0, stream>>>(
        c, Wkv + (size_t)l * DMOD * 2 * DMOD, kvb, nullptr);
    // q = h @ Wq[l]             (1024 x 1024 x 1024)
    gemm_bf16_wmma<0, DMOD, DMOD><<<dim3(XB / GBM, DMOD / GBN), 128, 0, stream>>>(
        h, Wq + (size_t)l * DMOD * DMOD, qb, nullptr);
    // r at the 4 surviving relative positions
    rel_kernel<<<dim3(MSEG + 1, DMOD / 256), 256, 0, stream>>>(
        Wrel + (size_t)l * DMOD * DMOD, r4);
    // diagonal scores + softmax(S=2) + V gather
    attn_kernel<<<XLEN * BNUM * HNUM, 64, 0, stream>>>(
        qb, kvb, r4, u_bias + (size_t)l * HNUM * DHEAD,
        v_bias + (size_t)l * HNUM * DHEAD, indices, avb);
    // attn = av @ Wo[l]
    gemm_bf16_wmma<0, DMOD, DMOD><<<dim3(XB / GBM, DMOD / GBN), 128, 0, stream>>>(
        avb, Wo + (size_t)l * DMOD * DMOD, attn, nullptr);
    // h = LN(h + attn)
    ln_residual_kernel<<<XB, 256, 0, stream>>>(h, attn, ln1g + (size_t)l * DMOD,
                                               ln1b + (size_t)l * DMOD);
    // hid = relu(h @ W1[l] + b1) (1024 x 4096 x 1024)
    gemm_bf16_wmma<1, DFFN, DMOD><<<dim3(XB / GBM, DFFN / GBN), 128, 0, stream>>>(
        h, W1 + (size_t)l * DMOD * DFFN, hid, b1f + (size_t)l * DFFN);
    // ff = hid @ W2[l] + b2      (1024 x 1024 x 4096)
    gemm_bf16_wmma<2, DMOD, DFFN><<<dim3(XB / GBM, DMOD / GBN), 128, 0, stream>>>(
        hid, W2 + (size_t)l * DFFN * DMOD, ffb, b2f + (size_t)l * DMOD);
    // h = LN(h + ff)
    ln_residual_kernel<<<XB, 256, 0, stream>>>(h, ffb, ln2g + (size_t)l * DMOD,
                                               ln2b + (size_t)l * DMOD);
  }

  // logits = h @ dec_w          (1024 x 32000 x 1024)
  gemm_bf16_wmma<0, VOC, DMOD><<<dim3(XB / GBM, VOC / GBN), 128, 0, stream>>>(
      h, dec_w, (float*)d_out, nullptr);
}
